// AFNO2D_16862041604096
// MI455X (gfx1250) — matrix operand order repaired and checked by reading, not compiled, not run on hardware
//
#include <hip/hip_runtime.h>
#include <stdint.h>

typedef __attribute__((ext_vector_type(16))) _Float16 v16h;
typedef __attribute__((ext_vector_type(4)))  _Float16 v4h;
typedef __attribute__((ext_vector_type(8)))  float    v8f;
typedef __attribute__((ext_vector_type(4)))  int      v4i;

#define MROWS 32768   // B*N = 8*4096 rows
#define CDIM  768     // channels
#define NB    8       // blocks
#define BS    96      // block size

#define AS1 __attribute__((address_space(1)))
#define AS3 __attribute__((address_space(3)))

#if defined(__AMDGCN__) && __has_builtin(__builtin_amdgcn_global_load_async_to_lds_b128)
#define HAVE_ASYNC 1
#else
#define HAVE_ASYNC 0
#endif

// 16-byte global -> LDS copy; async (ASYNCcnt-tracked) when available.
static __device__ __forceinline__ void ld16(void* l, const void* g) {
#if HAVE_ASYNC
  __builtin_amdgcn_global_load_async_to_lds_b128(
      (AS1 v4i*)(uintptr_t)g,
      (AS3 v4i*)(uint32_t)(uintptr_t)l, 0, 0);
#else
  *(float4*)l = *(const float4*)g;
#endif
}

static __device__ __forceinline__ void wait_async() {
#if HAVE_ASYNC
#if __has_builtin(__builtin_amdgcn_s_wait_asynccnt)
  __builtin_amdgcn_s_wait_asynccnt(0);
#else
  asm volatile("s_wait_asynccnt 0x0" ::: "memory");
#endif
#endif
}

static __device__ __forceinline__ v8f wmma_f16(v16h a, v16h b, v8f c) {
  // D = A(16x32 f16) * B(32x16 f16) + C(16x16 f32)
  return __builtin_amdgcn_wmma_f32_16x16x32_f16(false, a, false, b, (short)0, c,
                                                false, false);
}

// A-fragment permutation: lane half0 wants K {0..7,16..23} contiguous, half1
// wants {8..15,24..31} => swap k-groups 1 and 2 (16-byte granules).
static __device__ __forceinline__ int gmap(int g) {
  return (g == 1 || g == 2) ? (g ^ 3) : g;
}
static __device__ __forceinline__ int kslot(int k) {
  return gmap(k >> 3) * 8 + (k & 7);
}

// ---------------------------------------------------------------------------
// cas[i][j] = cos(2*pi*i*j/C) + sin(2*pi*i*j/C), computed via (i*j) mod C.
__global__ void build_cas(_Float16* __restrict__ cas) {
  int idx = blockIdx.x * blockDim.x + threadIdx.x;
  if (idx >= CDIM * CDIM) return;
  int i = idx / CDIM, j = idx % CDIM;
  int m = (i * j) % CDIM;
  float ang = 6.28318530717958647692f * (float)m * (1.0f / (float)CDIM);
  cas[idx] = (_Float16)(cosf(ang) + sinf(ang));
}

// f32 -> f16 conversion, 4 elements per thread.
__global__ void conv_f32_f16(const float* __restrict__ src,
                             _Float16* __restrict__ dst, int n4) {
  int i = blockIdx.x * blockDim.x + threadIdx.x;
  if (i >= n4) return;
  float4 v = ((const float4*)src)[i];
  v4h o = {(_Float16)v.x, (_Float16)v.y, (_Float16)v.z, (_Float16)v.w};
  ((v4h*)dst)[i] = o;
}

// ---------------------------------------------------------------------------
// C = A[M x 768] * cas[768 x 768]; cas is SYMMETRIC so B columns == B rows.
// 256 threads = 8 waves; tile 128x128; wave w owns rows w*16..w*16+15.
// Double-buffered async pipeline: stage tile k+1 while WMMAs run on tile k.
// FINAL=0: store f16 to Yout.  FINAL=1: Fout = acc*scale + Xbias (f32).
template <int FINAL>
__global__ void gemm_cas(const _Float16* __restrict__ A,
                         const _Float16* __restrict__ Bm,
                         _Float16* __restrict__ Yout, float* __restrict__ Fout,
                         const float* __restrict__ Xbias, float scale) {
  __shared__ __align__(32) _Float16 lds_a[2][128 * 32];
  __shared__ __align__(32) _Float16 lds_b[2][128 * 32];
  const int tid = threadIdx.x;
  const int lane = tid & 31, wave = tid >> 5;
  const int r = lane & 15, half = lane >> 4;
  const int m0 = blockIdx.y * 128, n0 = blockIdx.x * 128;

  // each thread stages 2 granules of A and 2 of B per k-step
  const int srow = tid >> 2, sg = tid & 3;

  auto stage = [&](int buf, int k0) {
#pragma unroll
    for (int i = 0; i < 2; ++i) {
      int row = srow + i * 64, g = sg;
      ld16((char*)lds_a[buf] + row * 64 + gmap(g) * 16,
           A + (size_t)(m0 + row) * CDIM + k0 + g * 8);
      ld16((char*)lds_b[buf] + row * 64 + gmap(g) * 16,
           Bm + (size_t)(n0 + row) * CDIM + k0 + g * 8);
    }
  };

  v8f acc[8];
  const v8f vzero = {0.f, 0.f, 0.f, 0.f, 0.f, 0.f, 0.f, 0.f};
#pragma unroll
  for (int j = 0; j < 8; ++j) acc[j] = vzero;

  stage(0, 0);
  wait_async();
  __syncthreads();

  for (int kt = 0; kt < CDIM / 32; ++kt) {
    const int cur = kt & 1;
    if (kt < CDIM / 32 - 1) stage(cur ^ 1, (kt + 1) * 32);  // async prefetch
    v16h af = *(const v16h*)(lds_a[cur] + (wave * 16 + r) * 32 + half * 16);
#pragma unroll
    for (int j = 0; j < 8; ++j) {
      v16h bf = *(const v16h*)(lds_b[cur] + (j * 16 + r) * 32 + half * 16);
      acc[j] = wmma_f16(af, bf, acc[j]);
    }
    if (kt < CDIM / 32 - 1) {
      wait_async();        // next tile landed
      __syncthreads();     // everyone done reading cur, next tile visible
    }
  }

  // D layout: row = vgpr + 8*half, col = tile*16 + (lane&15)
  const int rbase = m0 + wave * 16 + half * 8;
#pragma unroll
  for (int j = 0; j < 8; ++j) {
    int ncol = n0 + j * 16 + r;
#pragma unroll
    for (int v = 0; v < 8; ++v) {
      size_t off = (size_t)(rbase + v) * CDIM + ncol;
      if (FINAL)
        Fout[off] = acc[j][v] * scale + Xbias[off];
      else
        Yout[off] = (_Float16)acc[j][v];
    }
  }
}

// ---------------------------------------------------------------------------
// Fused block-diagonal complex MLP + softshrink + (real - imag).
// 128 threads = 4 waves; tile = 64 rows of one 96-channel block.
__global__ void afno_mlp(const _Float16* __restrict__ Y, _Float16* __restrict__ Z,
                         const _Float16* __restrict__ w1h,
                         const _Float16* __restrict__ b1h,
                         const _Float16* __restrict__ w2h,
                         const _Float16* __restrict__ b2h) {
  __shared__ __align__(32) _Float16 lds_y[64 * 96];    // A-tile layer1 (permuted)
  __shared__ __align__(32) _Float16 lds_o1[64 * 192];  // o1r|o1i (permuted)
  __shared__ __align__(32) _Float16 wbuf[96 * 96];     // one weight, [out][in]
  const int tid = threadIdx.x;
  const int lane = tid & 31, wave = tid >> 5;
  const int r = lane & 15, half = lane >> 4;
  const int m0 = blockIdx.x * 64;
  const int kb = blockIdx.y;

  const _Float16* Ybase = Y + (size_t)m0 * CDIM + kb * BS;
  _Float16* Zbase = Z + (size_t)m0 * CDIM + kb * BS;

  // ---- stage Y tile: 64 x 96 f16, group-permuted, 16-byte async granules.
  // 64 rows * 12 granules = 768; 6 per thread. (not unrolled: short addr lifetimes)
#pragma unroll 1
  for (int i = 0; i < 6; ++i) {
    int p = tid + i * 128;
    int row = p / 12, cw = p % 12;
    int win = cw >> 2, g = cw & 3;
    ld16((char*)lds_y + row * 192 + win * 64 + gmap(g) * 16,
         Ybase + row * CDIM + win * 32 + g * 8);
  }

  // stage one 96x96 weight (row-major [in][out]) transposed into wbuf[out][in]
  auto stageW = [&](const _Float16* src) {
#pragma unroll 1
    for (int i = 0; i < 36; ++i) {
      int p = tid + i * 128;          // 4608 dword-pairs
      int ii = p / 48, oo = (p % 48) * 2;
      union { uint32_t u; _Float16 h[2]; } u;
      u.u = *(const uint32_t*)(src + ii * BS + oo);
      wbuf[(oo + 0) * BS + (ii >> 5) * 32 + kslot(ii & 31)] = u.h[0];
      wbuf[(oo + 1) * BS + (ii >> 5) * 32 + kslot(ii & 31)] = u.h[1];
    }
  };

  // layer1: acc[j] += lds_y(16x96) @ wbuf(96 x 16-col-tile j)
  auto gemm1 = [&](v8f* acc) {
#pragma unroll
    for (int win = 0; win < 3; ++win) {
      v16h af = *(const v16h*)(lds_y + (wave * 16 + r) * 96 + win * 32 + half * 16);
#pragma unroll
      for (int j = 0; j < 6; ++j) {
        v16h bf = *(const v16h*)(wbuf + (j * 16 + r) * 96 + win * 32 + half * 16);
        acc[j] = wmma_f16(af, bf, acc[j]);
      }
    }
  };

  // bias + relu, write o1 into lds_o1 (cols br*96.., window-permuted)
  auto storeO1 = [&](const v8f* acc, int br) {
#pragma unroll
    for (int j = 0; j < 6; ++j) {
      float bv = (float)b1h[(br * NB + kb) * BS + j * 16 + r];
#pragma unroll
      for (int v = 0; v < 8; ++v) {
        float o = acc[j][v] + bv;
        o = o > 0.f ? o : 0.f;
        int row = wave * 16 + v + half * 8;
        int col = br * BS + j * 16 + r;
        lds_o1[row * 192 + ((col >> 5) * 32 + kslot(col & 31))] = (_Float16)o;
      }
    }
  };

  // layer2 bulk: acc[j] += o1(windows winBase..winBase+2) @ wbuf
  auto gemm2 = [&](v8f* acc, int winBase) {
#pragma unroll
    for (int w = 0; w < 3; ++w) {
      v16h af = *(const v16h*)(lds_o1 + (wave * 16 + r) * 192 +
                               (winBase + w) * 32 + half * 16);
#pragma unroll
      for (int j = 0; j < 6; ++j) {
        v16h bf = *(const v16h*)(wbuf + (j * 16 + r) * 96 + w * 32 + half * 16);
        acc[j] = wmma_f16(af, bf, acc[j]);
      }
    }
  };

  const _Float16* W1r = w1h + (size_t)(0 * NB + kb) * BS * BS;
  const _Float16* W1i = w1h + (size_t)(1 * NB + kb) * BS * BS;
  const _Float16* W2r = w2h + (size_t)(0 * NB + kb) * BS * BS;
  const _Float16* W2i = w2h + (size_t)(1 * NB + kb) * BS * BS;

  const v8f vzero = {0.f, 0.f, 0.f, 0.f, 0.f, 0.f, 0.f, 0.f};

  stageW(W1r);
  wait_async();               // lds_y granules landed
  __syncthreads();

  {                           // o1r: single accumulator array, freed via LDS
    v8f a[6];
#pragma unroll
    for (int j = 0; j < 6; ++j) a[j] = vzero;
    gemm1(a);
    storeO1(a, 0);
  }
  __syncthreads();            // done reading wbuf
  stageW(W1i);
  __syncthreads();
  {                           // o1i reuses the same register budget
    v8f a[6];
#pragma unroll
    for (int j = 0; j < 6; ++j) a[j] = vzero;
    gemm1(a);
    storeO1(a, 1);
  }
  __syncthreads();            // o1 visible; wbuf free

  v8f ar[6], ai[6];
#pragma unroll
  for (int j = 0; j < 6; ++j) { ar[j] = vzero; ai[j] = vzero; }

  stageW(W2r);
  __syncthreads();
  gemm2(ar, 0);               // o1r @ W2r
  gemm2(ai, 3);               // o1i @ W2r
  __syncthreads();
  stageW(W2i);
  __syncthreads();

  // Per-tile finish: tm is an 8-VGPR transient instead of a 48-VGPR array.
  const float lam = 0.01f;
#pragma unroll
  for (int j = 0; j < 6; ++j) {
    v8f tm = vzero;
#pragma unroll
    for (int w = 0; w < 3; ++w) {
      v16h bf  = *(const v16h*)(wbuf + (j * 16 + r) * 96 + w * 32 + half * 16);
      v16h afi = *(const v16h*)(lds_o1 + (wave * 16 + r) * 192 +
                                (3 + w) * 32 + half * 16);
      v16h afr = *(const v16h*)(lds_o1 + (wave * 16 + r) * 192 +
                                w * 32 + half * 16);
      tm = wmma_f16(afi, bf, tm);        // o1i @ W2i (to subtract)
      ai[j] = wmma_f16(afr, bf, ai[j]);  // o1r @ W2i
    }
    int col = j * 16 + r;
    float b2r = (float)b2h[(0 * NB + kb) * BS + col];
    float b2i = (float)b2h[(1 * NB + kb) * BS + col];
#pragma unroll
    for (int v = 0; v < 8; ++v) {
      float o2r = ar[j][v] - tm[v] + b2r;
      float o2i = ai[j][v] + b2i;
      float s1 = fabsf(o2r) - lam; s1 = s1 > 0.f ? s1 : 0.f; s1 = copysignf(s1, o2r);
      float s2 = fabsf(o2i) - lam; s2 = s2 > 0.f ? s2 : 0.f; s2 = copysignf(s2, o2i);
      int row = wave * 16 + v + half * 8;
      Zbase[row * CDIM + col] = (_Float16)(s1 - s2);
    }
  }
}

// ---------------------------------------------------------------------------
extern "C" void kernel_launch(void* const* d_in, const int* in_sizes, int n_in,
                              void* d_out, int out_size, void* d_ws, size_t ws_size,
                              hipStream_t stream) {
  const float* x  = (const float*)d_in[0];
  const float* w1 = (const float*)d_in[1];
  const float* b1 = (const float*)d_in[2];
  const float* w2 = (const float*)d_in[3];
  const float* b2 = (const float*)d_in[4];
  float* out = (float*)d_out;

  char* ws = (char*)d_ws;
  size_t off = 0;
  _Float16* cas_h = (_Float16*)(ws + off); off += (size_t)CDIM * CDIM * 2;
  _Float16* XhZ   = (_Float16*)(ws + off); off += (size_t)MROWS * CDIM * 2; // Xh, later Z
  _Float16* Yb    = (_Float16*)(ws + off); off += (size_t)MROWS * CDIM * 2;
  _Float16* w1h   = (_Float16*)(ws + off); off += (size_t)2 * NB * BS * BS * 2;
  _Float16* b1h   = (_Float16*)(ws + off); off += (size_t)2 * NB * BS * 2;
  _Float16* w2h   = (_Float16*)(ws + off); off += (size_t)2 * NB * BS * BS * 2;
  _Float16* b2h   = (_Float16*)(ws + off);

  build_cas<<<(CDIM * CDIM + 255) / 256, 256, 0, stream>>>(cas_h);
  conv_f32_f16<<<(MROWS * CDIM / 4 + 255) / 256, 256, 0, stream>>>(x, XhZ, MROWS * CDIM / 4);
  conv_f32_f16<<<(2 * NB * BS * BS / 4 + 255) / 256, 256, 0, stream>>>(w1, w1h, 2 * NB * BS * BS / 4);
  conv_f32_f16<<<(2 * NB * BS / 4 + 255) / 256, 256, 0, stream>>>(b1, b1h, 2 * NB * BS / 4);
  conv_f32_f16<<<(2 * NB * BS * BS / 4 + 255) / 256, 256, 0, stream>>>(w2, w2h, 2 * NB * BS * BS / 4);
  conv_f32_f16<<<(2 * NB * BS / 4 + 255) / 256, 256, 0, stream>>>(b2, b2h, 2 * NB * BS / 4);

  dim3 gg(CDIM / 128, MROWS / 128);
  // Y = Xh @ cas
  gemm_cas<0><<<gg, 256, 0, stream>>>(XhZ, cas_h, Yb, nullptr, nullptr, 1.0f);
  // Z = softshrink-real - softshrink-imag of block MLP (overwrites Xh buffer)
  afno_mlp<<<dim3(MROWS / 64, NB), 128, 0, stream>>>(Yb, XhZ, w1h, b1h, w2h, b2h);
  // out = (Z @ cas) / n_tot + x
  const float scale = 1.0f / (float)((size_t)8 * 64 * 64 * 768);
  gemm_cas<1><<<gg, 256, 0, stream>>>(XhZ, cas_h, nullptr, out, x, scale);
}
